// CrystalGCNEncoder_16260746183174
// MI455X (gfx1250) — compile-verified, matrix operand order checked by reading, not run on hardware
//
#include <hip/hip_runtime.h>
#include <hip/hip_bf16.h>

typedef unsigned short ushort_t;
typedef __attribute__((ext_vector_type(16))) __bf16 v16bf;
typedef __attribute__((ext_vector_type(8)))  float  v8f;

struct __attribute__((aligned(16))) U128 { unsigned x, y, z, w; };
union ABu { v16bf v; U128 q[2]; };

#define ENC_NEG_INF 0x007FFFFFu  // order-preserving encoding of -inf

__device__ __forceinline__ unsigned fenc(float f) {
    unsigned u = __float_as_uint(f);
    return (u & 0x80000000u) ? ~u : (u | 0x80000000u);
}
__device__ __forceinline__ float fdec(unsigned u) {
    unsigned v = (u & 0x80000000u) ? (u & 0x7FFFFFFFu) : ~u;
    return __uint_as_float(v);
}
__device__ __forceinline__ ushort_t f32_to_bf16(float f) {
    unsigned u = __float_as_uint(f);
    u += 0x7FFFu + ((u >> 16) & 1u);   // round-to-nearest-even
    return (ushort_t)(u >> 16);
}

// ---------------- generic fills / converts ----------------
__global__ void fill_u32(unsigned* p, unsigned v, int n) {
    int i = blockIdx.x * blockDim.x + threadIdx.x;
    if (i < n) p[i] = v;
}
__global__ void cvt_bf16(const float* x, ushort_t* y, int n) {
    int i = blockIdx.x * blockDim.x + threadIdx.x;
    if (i < n) y[i] = f32_to_bf16(x[i]);
}
// broadcast bias row into every node row (out = b, then scatter-add on top)
__global__ void init_bias(float* out, const float* b, int total, int F) {
    int i = blockIdx.x * blockDim.x + threadIdx.x;
    if (i < total) out[i] = b[i % F];
}
__global__ void elu_ip(float* x, int n) {
    int i = blockIdx.x * blockDim.x + threadIdx.x;
    if (i < n) { float v = x[i]; x[i] = v > 0.f ? v : expm1f(v); }
}

// ---------------- weight repack: fragment-friendly bf16 ----------------
// Packed layout: ((nt*KT + kt)*32 + lane)*16 + e ; lane e-th half is
// W[k, n] with n = nt*16 + (lane&15), k = kt*32 + 8*(lane>>4) + (e<8 ? e : e+8)
__global__ void pack_w_bf16(const float* W, ushort_t* out, int K, int Nout) {
    int idx = blockIdx.x * blockDim.x + threadIdx.x;
    int total = K * Nout;
    if (idx >= total) return;
    int e    = idx & 15;
    int lane = (idx >> 4) & 31;
    int t    = idx >> 9;
    int KT   = K >> 5;
    int kt   = t % KT;
    int nt   = t / KT;
    int n    = nt * 16 + (lane & 15);
    int k    = kt * 32 + ((lane >> 4) << 3) + (e < 8 ? e : e + 8);
    out[idx] = f32_to_bf16(W[(size_t)k * Nout + n]);
}

// ---------------- WMMA GEMM: C(MxN) = A(MxK)@B(KxN) [+bias][relu] ----------------
// A: bf16 row-major; Bp: packed tiles (above); C: f32 row-major.
// block = 256 threads = 8 waves; each wave owns one 16x16 output tile.
__global__ __launch_bounds__(256)
void gemm_wmma(const ushort_t* A, const ushort_t* Bp, const float* bias,
               float* C, int K, int Nout, int act) {
    int lane = threadIdx.x & 31;
    int wave = threadIdx.x >> 5;
    int rowTile = blockIdx.x;
    int nt = blockIdx.y * 8 + wave;
    int KT = K >> 5;
    int r  = lane & 15;
    int hs = lane >> 4;

    v8f acc = {};
    const ushort_t* arow  = A + (size_t)(rowTile * 16 + r) * K + hs * 8;
    const ushort_t* btile = Bp + ((size_t)nt * KT * 32 + lane) * 16;

    for (int kt = 0; kt < KT; ++kt) {
        ABu a, b;
        a.q[0] = *(const U128*)(arow);
        a.q[1] = *(const U128*)(arow + 16);
        b.q[0] = *(const U128*)(btile);
        b.q[1] = *(const U128*)(btile + 8);
        acc = __builtin_amdgcn_wmma_f32_16x16x32_bf16(
            false, a.v, false, b.v, (short)0, acc, false, false);
        arow  += 32;
        btile += 32 * 16;
    }

    int col = nt * 16 + r;
    float bv = bias ? bias[col] : 0.f;
    int rbase = rowTile * 16 + hs * 8;
    #pragma unroll
    for (int v = 0; v < 8; ++v) {
        float val = acc[v] + bv;
        if (act == 1) val = val > 0.f ? val : 0.f;
        C[(size_t)(rbase + v) * Nout + col] = val;
    }
}

// ---------------- tiny K GEMM (layer 1: K=4) ----------------
__global__ void gemm_small(const float* x, const float* W, float* out,
                           int Nn, int Fin, int Fout) {
    int idx = blockIdx.x * blockDim.x + threadIdx.x;
    if (idx >= Nn * Fout) return;
    int n = idx / Fout, j = idx % Fout;
    float s = 0.f;
    for (int k = 0; k < Fin; ++k) s += x[(size_t)n * Fin + k] * W[(size_t)k * Fout + j];
    out[idx] = s;
}

// ---------------- GAT attention pipeline ----------------
__global__ void attn_scores(const float* h, const float* al, const float* ar,
                            float* el, float* er, int Nn, int H, int D) {
    int idx = blockIdx.x * blockDim.x + threadIdx.x;
    if (idx >= Nn * H) return;
    int n = idx / H, hh = idx % H;
    const float* hp = h + (size_t)n * H * D + (size_t)hh * D;
    float sl = 0.f, sr = 0.f;
    for (int d = 0; d < D; ++d) { float v = hp[d]; sl += v * al[hh * D + d]; sr += v * ar[hh * D + d]; }
    el[idx] = sl; er[idx] = sr;
}

__global__ void edge_max(const float* el, const float* er, const int* src,
                         const int* dst, unsigned* menc, int En, int H) {
    int idx = blockIdx.x * blockDim.x + threadIdx.x;
    if (idx >= En * H) return;
    int i = idx / H, hh = idx % H;
    float e = el[src[i] * H + hh] + er[dst[i] * H + hh];
    e = e > 0.f ? e : 0.2f * e;
    atomicMax(&menc[dst[i] * H + hh], fenc(e));
}

__global__ void edge_exp(const float* el, const float* er, const int* src,
                         const int* dst, const unsigned* menc, float* alpha,
                         float* ssum, int En, int H) {
    int idx = blockIdx.x * blockDim.x + threadIdx.x;
    if (idx >= En * H) return;
    int i = idx / H, hh = idx % H;
    int d = dst[i];
    float e = el[src[i] * H + hh] + er[d * H + hh];
    e = e > 0.f ? e : 0.2f * e;
    float ex = __expf(e - fdec(menc[d * H + hh]));
    alpha[idx] = ex;
    atomicAdd(&ssum[d * H + hh], ex);
}

__global__ void edge_norm(float* alpha, const int* dst, const float* ssum,
                          int En, int H) {
    int idx = blockIdx.x * blockDim.x + threadIdx.x;
    if (idx >= En * H) return;
    int i = idx / H, hh = idx % H;
    alpha[idx] = alpha[idx] / ssum[dst[i] * H + hh];
}

// heavy kernel: per-edge gather h[src]*alpha, atomic scatter-add into out[dst]
__global__ __launch_bounds__(256)
void edge_msg(const float* h, const float* alpha, const int* src,
              const int* dst, float* out, int H, int D) {
    int i = blockIdx.x;
    int F = H * D;
    int si = src[i], di = dst[i];
    const float* hp = h + (size_t)si * F;
    float* op = out + (size_t)di * F;
    for (int f = threadIdx.x; f < F; f += blockDim.x) {
        float a = alpha[(size_t)i * H + f / D];
        atomicAdd(&op[f], hp[f] * a);
    }
}

// ---------------- pooling + heads ----------------
__global__ void gate_dot(const float* g1, const float* w, const float* b,
                         float* gate, int Nn, int Kd) {
    int lane = threadIdx.x & 31;
    int wave = threadIdx.x >> 5;
    int n = blockIdx.x * 8 + wave;
    if (n >= Nn) return;
    float s = 0.f;
    for (int k = lane; k < Kd; k += 32) s += g1[(size_t)n * Kd + k] * w[k];
    for (int off = 16; off > 0; off >>= 1) s += __shfl_down(s, off, 32);
    if (lane == 0) gate[n] = s + b[0];
}

__global__ void gmax_k(const float* gate, const int* gid, unsigned* genc, int Nn) {
    int n = blockIdx.x * blockDim.x + threadIdx.x;
    if (n < Nn) atomicMax(&genc[gid[n]], fenc(gate[n]));
}
__global__ void gexp_k(const float* gate, const int* gid, const unsigned* genc,
                       float* ge, float* gsum, int Nn) {
    int n = blockIdx.x * blockDim.x + threadIdx.x;
    if (n >= Nn) return;
    float ex = __expf(gate[n] - fdec(genc[gid[n]]));
    ge[n] = ex;
    atomicAdd(&gsum[gid[n]], ex);
}
__global__ void gnorm_k(const float* ge, const int* gid, const float* gsum,
                        float* an, int Nn) {
    int n = blockIdx.x * blockDim.x + threadIdx.x;
    if (n < Nn) an[n] = ge[n] / gsum[gid[n]];
}
__global__ __launch_bounds__(256)
void pool_k(const float* h, const float* an, const int* gid, float* gemb, int F) {
    int n = blockIdx.x;
    float a = an[n];
    int g = gid[n];
    const float* hp = h + (size_t)n * F;
    float* gp = gemb + (size_t)g * F;
    for (int f = threadIdx.x; f < F; f += blockDim.x) atomicAdd(&gp[f], a * hp[f]);
}
__global__ void heads_k(const float* gemb, const float* Wmu, const float* bmu,
                        const float* Wlv, const float* blv, float* out,
                        int Gn, int F, int L) {
    int idx = blockIdx.x * blockDim.x + threadIdx.x;
    if (idx >= Gn * L) return;
    int g = idx / L, j = idx % L;
    const float* ep = gemb + (size_t)g * F;
    float smu = 0.f, slv = 0.f;
    for (int k = 0; k < F; ++k) {
        float e = ep[k];
        smu += e * Wmu[(size_t)k * L + j];
        slv += e * Wlv[(size_t)k * L + j];
    }
    out[idx] = smu + bmu[j];
    out[(size_t)Gn * L + idx] = slv + blv[j];
}

// ---------------- driver ----------------
extern "C" void kernel_launch(void* const* d_in, const int* in_sizes, int n_in,
                              void* d_out, int out_size, void* d_ws, size_t ws_size,
                              hipStream_t stream) {
    (void)in_sizes; (void)n_in; (void)out_size; (void)ws_size;
    const float* feat = (const float*)d_in[0];
    const int*   src  = (const int*)d_in[1];
    const int*   dst  = (const int*)d_in[2];
    const int*   gid  = (const int*)d_in[3];
    const float* Ws[3]  = {(const float*)d_in[4],  (const float*)d_in[8],  (const float*)d_in[12]};
    const float* als[3] = {(const float*)d_in[5],  (const float*)d_in[9],  (const float*)d_in[13]};
    const float* ars[3] = {(const float*)d_in[6],  (const float*)d_in[10], (const float*)d_in[14]};
    const float* bs[3]  = {(const float*)d_in[7],  (const float*)d_in[11], (const float*)d_in[15]};
    const float* Wg1 = (const float*)d_in[16];
    const float* bg1 = (const float*)d_in[17];
    const float* Wg2 = (const float*)d_in[18];
    const float* bg2 = (const float*)d_in[19];
    const float* Wmu = (const float*)d_in[20];
    const float* bmu = (const float*)d_in[21];
    const float* Wlv = (const float*)d_in[22];
    const float* blv = (const float*)d_in[23];

    const int N = 10000, E = 160000, G = 50, H = 4;
    const int Ds[3]   = {64, 128, 256};
    const int Fins[3] = {4, 256, 512};

    char* w = (char*)d_ws;
    size_t off = 0;
    auto alloc = [&](size_t bytes) -> void* {
        void* p = w + off;
        off = (off + bytes + 255) & ~(size_t)255;
        return p;
    };
    float*    h1    = (float*)alloc((size_t)N * 256 * 4);
    float*    h2    = (float*)alloc((size_t)N * 512 * 4);
    float*    h3    = (float*)alloc((size_t)N * 1024 * 4);
    float*    htmp  = (float*)alloc((size_t)N * 1024 * 4);   // x@W, later gate hidden
    ushort_t* hb    = (ushort_t*)alloc((size_t)N * 1024 * 2);
    ushort_t* wp    = (ushort_t*)alloc((size_t)512 * 1024 * 2);
    float*    el    = (float*)alloc((size_t)N * H * 4);
    float*    er    = (float*)alloc((size_t)N * H * 4);
    unsigned* menc  = (unsigned*)alloc((size_t)N * H * 4);
    float*    ssum  = (float*)alloc((size_t)N * H * 4);
    float*    alpha = (float*)alloc((size_t)E * H * 4);
    float*    gatev = (float*)alloc((size_t)N * 4);
    float*    ge    = (float*)alloc((size_t)N * 4);
    float*    an    = (float*)alloc((size_t)N * 4);
    unsigned* genc  = (unsigned*)alloc((size_t)G * 4);
    float*    gsum  = (float*)alloc((size_t)G * 4);
    float*    gemb  = (float*)alloc((size_t)G * 1024 * 4);

    float* outs[3] = {h1, h2, h3};
    const float* xin = feat;
    const int TB = 256;
    auto blocks = [](int n, int b) { return (n + b - 1) / b; };

    for (int l = 0; l < 3; ++l) {
        int D = Ds[l], F = H * D, Fin = Fins[l];
        // h = x @ W
        if (l == 0) {
            gemm_small<<<blocks(N * F, TB), TB, 0, stream>>>(xin, Ws[l], htmp, N, Fin, F);
        } else {
            cvt_bf16<<<blocks(N * Fin, TB), TB, 0, stream>>>(xin, hb, N * Fin);
            pack_w_bf16<<<blocks(Fin * F, TB), TB, 0, stream>>>(Ws[l], wp, Fin, F);
            gemm_wmma<<<dim3(N / 16, F / 128), TB, 0, stream>>>(hb, wp, nullptr, htmp, Fin, F, 0);
        }
        // attention scores + edge softmax over dst
        attn_scores<<<blocks(N * H, TB), TB, 0, stream>>>(htmp, als[l], ars[l], el, er, N, H, D);
        fill_u32<<<blocks(N * H, TB), TB, 0, stream>>>(menc, ENC_NEG_INF, N * H);
        fill_u32<<<blocks(N * H, TB), TB, 0, stream>>>((unsigned*)ssum, 0u, N * H);
        edge_max<<<blocks(E * H, TB), TB, 0, stream>>>(el, er, src, dst, menc, E, H);
        edge_exp<<<blocks(E * H, TB), TB, 0, stream>>>(el, er, src, dst, menc, alpha, ssum, E, H);
        edge_norm<<<blocks(E * H, TB), TB, 0, stream>>>(alpha, dst, ssum, E, H);
        // aggregate: out = bias; out[dst] += h[src] * alpha; elu
        init_bias<<<blocks(N * F, TB), TB, 0, stream>>>(outs[l], bs[l], N * F, F);
        edge_msg<<<E, TB, 0, stream>>>(htmp, alpha, src, dst, outs[l], H, D);
        elu_ip<<<blocks(N * F, TB), TB, 0, stream>>>(outs[l], N * F);
        xin = outs[l];
    }

    // gate MLP: relu(h3 @ Wg1 + bg1) @ Wg2 + bg2
    cvt_bf16<<<blocks(N * 1024, TB), TB, 0, stream>>>(h3, hb, N * 1024);
    pack_w_bf16<<<blocks(1024 * 128, TB), TB, 0, stream>>>(Wg1, wp, 1024, 128);
    gemm_wmma<<<dim3(N / 16, 1), TB, 0, stream>>>(hb, wp, bg1, htmp, 1024, 128, 1);
    gate_dot<<<blocks(N, 8), TB, 0, stream>>>(htmp, Wg2, bg2, gatev, N, 128);

    // per-graph softmax + weighted pooling
    fill_u32<<<1, G, 0, stream>>>(genc, ENC_NEG_INF, G);
    fill_u32<<<1, G, 0, stream>>>((unsigned*)gsum, 0u, G);
    fill_u32<<<blocks(G * 1024, TB), TB, 0, stream>>>((unsigned*)gemb, 0u, G * 1024);
    gmax_k<<<blocks(N, TB), TB, 0, stream>>>(gatev, gid, genc, N);
    gexp_k<<<blocks(N, TB), TB, 0, stream>>>(gatev, gid, genc, ge, gsum, N);
    gnorm_k<<<blocks(N, TB), TB, 0, stream>>>(ge, gid, gsum, an, N);
    pool_k<<<N, TB, 0, stream>>>(h3, an, gid, gemb, 1024);

    // mu / logvar heads
    heads_k<<<blocks(G * 64, TB), TB, 0, stream>>>(gemb, Wmu, bmu, Wlv, blv,
                                                   (float*)d_out, G, 1024, 64);
}
